// LikelihoodEstimator_16630113370476
// MI455X (gfx1250) — matrix operand order repaired and checked by reading, not compiled, not run on hardware
//
#include <hip/hip_runtime.h>
#include <cmath>

// ---------------- problem constants ----------------
constexpr int PSZ   = 160;           // patch size
constexpr int NB    = 4;             // batch
constexpr int NC    = 3;             // channels
constexpr int NPIX  = PSZ * PSZ;     // 25600
constexpr int ND    = NC * NPIX;     // 76800 elems per image
constexpr int TILESX = PSZ / 16;     // 10 tiles per row
constexpr int NTILES = NB * PSZ * TILESX;  // 6400 waves of work
constexpr float T_EPS = 1e-5f;

typedef _Float16 v16h __attribute__((ext_vector_type(16)));
typedef _Float16 v8h  __attribute__((ext_vector_type(8)));
typedef float    v8f  __attribute__((ext_vector_type(8)));

__device__ __forceinline__ v8f wmma16(v16h a, v16h b, v8f c) {
    // D = A(16x32 f16) * B(32x16 f16) + C(16x16 f32)
    return __builtin_amdgcn_wmma_f32_16x16x32_f16(false, a, false, b,
                                                  (short)0, c, false, false);
}

// Build an A-fragment (16 pixels x 32 channels at one tap) from NHWC f16
// activations. Per ISA: lane<16 holds K=0..7 and 16..23; lane>=16 holds
// K=8..15 and 24..31.  Out-of-image -> zeros (SAME padding).
__device__ __forceinline__ v16h load_act_frag(const _Float16* __restrict__ act,
                                              int b, int yy, int xx, int lane) {
    v16h a = {};
    if ((unsigned)yy < (unsigned)PSZ && (unsigned)xx < (unsigned)PSZ) {
        const _Float16* p = act + (((size_t)((b * PSZ + yy) * PSZ + xx)) << 5)
                                + ((lane >> 4) << 3);
        v8h lo = *(const v8h*)p;          // channels kb..kb+7
        v8h hi = *(const v8h*)(p + 16);   // channels kb+16..kb+23
        a = __builtin_shufflevector(lo, hi, 0,1,2,3,4,5,6,7,8,9,10,11,12,13,14,15);
    }
    return a;
}

__device__ __forceinline__ void tile_decode(int tile, int& b, int& y, int& x0) {
    b = tile / (PSZ * TILESX);
    int rem = tile - b * (PSZ * TILESX);
    y = rem / TILESX;
    x0 = (rem - y * TILESX) * 16;
}

// ---------------- setup kernels ----------------

__global__ void init_kernel(const float* __restrict__ patches, float* __restrict__ xcur,
                            float* __restrict__ lp, float* __restrict__ lacc,
                            float* __restrict__ divb) {
    int id = blockIdx.x * blockDim.x + threadIdx.x;
    if (id < NB * ND) xcur[id] = patches[id];
    if (id < NB) { lp[id] = 0.f; lacc[id] = 0.f; divb[id] = 0.f; }
}

// Time embedding table fused with conv1 bias:
// embb[e][n] = b1[n] + bt[n] + (time MLP)(t_e)[n], e = step*4 + rk_stage.
__global__ void emb_kernel(const float* __restrict__ wt, const float* __restrict__ bt,
                           const float* __restrict__ b1, float* __restrict__ embb,
                           float dt) {
    int id = blockIdx.x * blockDim.x + threadIdx.x;
    if (id >= 40 * 32) return;
    int e = id >> 5, n = id & 31;
    const float ca[4] = {0.f, 0.5f, 0.5f, 1.f};
    float t = T_EPS + dt * (float)(e >> 2) + ca[e & 3] * dt;
    float c0 = -logf(10000.f) / 31.f;
    float s = bt[n] + b1[n];
    for (int k = 0; k < 32; ++k) {
        float f = expf(c0 * (float)k);
        float aa = t * 999.f * f;
        s += sinf(aa) * wt[k * 32 + n] + cosf(aa) * wt[(32 + k) * 32 + n];
    }
    embb[id] = s;
}

// u1 = conv1_nobias(eps), batch-independent, stored NHWC f16 [160][160][32]
__global__ void u1_kernel(const float* __restrict__ eps, const float* __restrict__ w1,
                          _Float16* __restrict__ u1) {
    int id = blockIdx.x * blockDim.x + threadIdx.x;
    if (id >= NPIX * 32) return;
    int o = id & 31; int pix = id >> 5; int y = pix / PSZ, x = pix - y * PSZ;
    float s = 0.f;
    for (int c = 0; c < 3; ++c)
        for (int ky = 0; ky < 3; ++ky)
            for (int kx = 0; kx < 3; ++kx) {
                int yy = y + ky - 1, xx = x + kx - 1;
                if ((unsigned)yy < (unsigned)PSZ && (unsigned)xx < (unsigned)PSZ)
                    s += eps[(c * PSZ + yy) * PSZ + xx] * w1[((o * 3 + c) * 3 + ky) * 3 + kx];
            }
    u1[id] = (_Float16)s;
}

// Pre-pack weight B-fragments in the per-lane ISA layout:
// lane<16: K = 0..7,16..23; lane>=16: K = 8..15,24..31.
// conv1: im2col k = tap*3 + c, padded 27->32.  pw1[ntile][lane] (2x32 v16h)
__global__ void pack_w1_kernel(const float* __restrict__ w1, v16h* __restrict__ pw1) {
    int id = threadIdx.x;
    if (id >= 2 * 32) return;
    int nt = id >> 5, lane = id & 31;
    int col = lane & 15, kb = (lane >> 4) * 8;
    v16h f = {};
    for (int j = 0; j < 16; ++j) {
        int k = (j < 8) ? kb + j : 16 + kb + (j - 8);
        if (k < 27) {
            int tap = k / 3, c = k - 3 * tap;
            f[j] = (_Float16)w1[((nt * 16 + col) * 3 + c) * 9 + tap];
        }
    }
    pw1[nt * 32 + lane] = f;
}

// conv2: per tap, K = 32 input channels.  pw2[ntile][tap][lane] (2x9x32 v16h)
__global__ void pack_w2_kernel(const float* __restrict__ w2, v16h* __restrict__ pw2) {
    int id = threadIdx.x;
    if (id >= 2 * 9 * 32) return;
    int nt = id / 288; int rem = id - nt * 288;
    int tap = rem >> 5, lane = rem & 31;
    int col = lane & 15, kb = (lane >> 4) * 8;
    v16h f;
    for (int j = 0; j < 16; ++j)
        f[j] = (_Float16)w2[((nt * 16 + col) * 32 + ((j < 8) ? kb + j : 16 + kb + (j - 8))) * 9 + tap];
    pw2[(nt * 9 + tap) * 32 + lane] = f;
}

// conv3: N padded 3->16.  pw3[tap][lane] (9x32 v16h)
__global__ void pack_w3_kernel(const float* __restrict__ w3, v16h* __restrict__ pw3) {
    int id = threadIdx.x;
    if (id >= 9 * 32) return;
    int tap = id >> 5, lane = id & 31;
    int col = lane & 15, kb = (lane >> 4) * 8;
    v16h f = {};
    if (col < 3)
        for (int j = 0; j < 16; ++j)
            f[j] = (_Float16)w3[(col * 32 + ((j < 8) ? kb + j : 16 + kb + (j - 8))) * 9 + tap];
    pw3[tap * 32 + lane] = f;
}

// ---------------- per-RHS conv kernels (WMMA) ----------------

// layer 1: z1 = conv1(x) + (b1+emb) ; h1 = relu(z1) ; th1 = (z1>0) ? u1 : 0
__global__ __launch_bounds__(256)
void conv1_kernel(const float* __restrict__ xin,
                  _Float16* __restrict__ h1, _Float16* __restrict__ th1,
                  const _Float16* __restrict__ u1, const v16h* __restrict__ pw1,
                  const float* __restrict__ embb, int eidx) {
    int lane = threadIdx.x & 31;
    int tile = blockIdx.x * 8 + (threadIdx.x >> 5);
    int b, y, x0; tile_decode(tile, b, y, x0);
    int px = x0 + (lane & 15);
    int kb = (lane >> 4) * 8;
    // A fragment: k = tap*3 + c (im2col, 27 valid), per ISA 16-bit A layout
    v16h a = {};
#pragma unroll
    for (int i = 0; i < 8; ++i) {
        int k0 = kb + i;            // <= 15, always a valid im2col index
        {
            int tap = k0 / 3, c = k0 - 3 * tap;
            int yy = y + tap / 3 - 1, xx = px + tap % 3 - 1;
            if ((unsigned)yy < (unsigned)PSZ && (unsigned)xx < (unsigned)PSZ)
                a[i] = (_Float16)xin[((b * NC + c) * PSZ + yy) * PSZ + xx];
        }
        int k1 = kb + 16 + i;
        if (k1 < 27) {
            int tap = k1 / 3, c = k1 - 3 * tap;
            int yy = y + tap / 3 - 1, xx = px + tap % 3 - 1;
            if ((unsigned)yy < (unsigned)PSZ && (unsigned)xx < (unsigned)PSZ)
                a[8 + i] = (_Float16)xin[((b * NC + c) * PSZ + yy) * PSZ + xx];
        }
    }
    v16h bw0 = pw1[lane], bw1 = pw1[32 + lane];
    v8f c0 = {}, c1 = {};
    c0 = wmma16(a, bw0, c0);
    c1 = wmma16(a, bw1, c1);

    int n = lane & 15, mb = (lane >> 4) << 3;
    // loop-invariant per lane: fused bias+embedding for both channel tiles
    float eb0 = embb[eidx * 32 + n];
    float eb1 = embb[eidx * 32 + 16 + n];
#pragma unroll
    for (int j = 0; j < 8; ++j) {
        int pix = x0 + mb + j;
        size_t base  = ((size_t)(b * PSZ + y) * PSZ + pix) << 5;
        size_t ubase = ((size_t)(y * PSZ + pix)) << 5;
        // unconditional u1 loads (always in-bounds), branch-free select
        _Float16 uv0 = u1[ubase + n];
        _Float16 uv1 = u1[ubase + 16 + n];
        float z0 = c0[j] + eb0;
        float z1 = c1[j] + eb1;
        h1[base + n]       = (_Float16)fmaxf(z0, 0.f);
        th1[base + n]      = (z0 > 0.f) ? uv0 : (_Float16)0.f;
        h1[base + 16 + n]  = (_Float16)fmaxf(z1, 0.f);
        th1[base + 16 + n] = (z1 > 0.f) ? uv1 : (_Float16)0.f;
    }
}

// layer 2: primal + tangent share weight fragments -> 4 WMMAs/tap
__global__ __launch_bounds__(256)
void conv2_kernel(const _Float16* __restrict__ h1, const _Float16* __restrict__ th1,
                  _Float16* __restrict__ h2, _Float16* __restrict__ th2,
                  const v16h* __restrict__ pw2, const float* __restrict__ b2) {
    int lane = threadIdx.x & 31;
    int tile = blockIdx.x * 8 + (threadIdx.x >> 5);
    int b, y, x0; tile_decode(tile, b, y, x0);
    int px = x0 + (lane & 15);
    v8f cp0 = {}, cp1 = {}, ct0 = {}, ct1 = {};
#pragma unroll
    for (int tap = 0; tap < 9; ++tap) {
        int yy = y + tap / 3 - 1, xx = px + tap % 3 - 1;
        v16h ap = load_act_frag(h1,  b, yy, xx, lane);
        v16h at = load_act_frag(th1, b, yy, xx, lane);
        v16h b0 = pw2[tap * 32 + lane];
        v16h b1w = pw2[(9 + tap) * 32 + lane];
        cp0 = wmma16(ap, b0,  cp0);
        cp1 = wmma16(ap, b1w, cp1);
        ct0 = wmma16(at, b0,  ct0);
        ct1 = wmma16(at, b1w, ct1);
    }
    int n = lane & 15, mb = (lane >> 4) << 3;
    float bb0 = b2[n], bb1 = b2[16 + n];
#pragma unroll
    for (int j = 0; j < 8; ++j) {
        int pix = x0 + mb + j;
        size_t base = ((size_t)(b * PSZ + y) * PSZ + pix) << 5;
        float z0 = cp0[j] + bb0;
        h2[base + n]       = (_Float16)fmaxf(z0, 0.f);
        th2[base + n]      = (z0 > 0.f) ? (_Float16)ct0[j] : (_Float16)0.f;
        float z1 = cp1[j] + bb1;
        h2[base + 16 + n]  = (_Float16)fmaxf(z1, 0.f);
        th2[base + 16 + n] = (z1 > 0.f) ? (_Float16)ct1[j] : (_Float16)0.f;
    }
}

// layer 3 + drift + Hutchinson dot:
//   k = -0.5*beta*x - cc*(conv3(h2)+b3),  div += -cc * sum(conv3(th2) * eps)
__global__ __launch_bounds__(256)
void conv3_drift_kernel(const _Float16* __restrict__ h2, const _Float16* __restrict__ th2,
                        const float* __restrict__ xin, const float* __restrict__ eps,
                        const v16h* __restrict__ pw3, const float* __restrict__ b3,
                        float* __restrict__ kbuf, float* __restrict__ divb,
                        float beta, float cc) {
    int lane = threadIdx.x & 31;
    int tile = blockIdx.x * 8 + (threadIdx.x >> 5);
    int b, y, x0; tile_decode(tile, b, y, x0);
    int px = x0 + (lane & 15);
    v8f cs = {}, ct = {};
#pragma unroll
    for (int tap = 0; tap < 9; ++tap) {
        int yy = y + tap / 3 - 1, xx = px + tap % 3 - 1;
        v16h ap = load_act_frag(h2,  b, yy, xx, lane);
        v16h at = load_act_frag(th2, b, yy, xx, lane);
        v16h bw = pw3[tap * 32 + lane];
        cs = wmma16(ap, bw, cs);
        ct = wmma16(at, bw, ct);
    }
    int n = lane & 15, mb = (lane >> 4) << 3;
    if (n < 3) {
        float bn = b3[n];
        float lsum = 0.f;
#pragma unroll
        for (int j = 0; j < 8; ++j) {
            int pix = x0 + mb + j;
            size_t idx = ((size_t)(b * NC + n) * PSZ + y) * PSZ + pix;
            float sv = cs[j] + bn;
            kbuf[idx] = -0.5f * beta * xin[idx] - cc * sv;
            float ev = eps[((size_t)n * PSZ + y) * PSZ + pix];
            lsum += ct[j] * ev;
        }
        atomicAdd(&divb[b], -cc * lsum);
    }
}

// ---------------- RK4 combine kernels ----------------

__global__ void combine_kernel(const float* __restrict__ xbase, const float* __restrict__ k,
                               float* __restrict__ kacc, float* __restrict__ xout,
                               float wk, float cdt, int stage, float dt6) {
    int id = blockIdx.x * blockDim.x + threadIdx.x;
    if (id >= NB * ND) return;
    float kv = k[id];
    float acc = (stage == 0 ? 0.f : kacc[id]) + wk * kv;
    kacc[id] = acc;
    if (stage < 3) xout[id] = xbase[id] + cdt * kv;
    else           xout[id] = xbase[id] + dt6 * acc;
}

__global__ void scalar_kernel(float* __restrict__ divb, float* __restrict__ lacc,
                              float* __restrict__ lp, float beta, float wk,
                              int stage, float dt6) {
    int b = threadIdx.x;
    if (b < NB) {
        float l = divb[b] - 0.5f * beta * (float)ND;  // eps^2 == 1 term
        float acc = (stage == 0 ? 0.f : lacc[b]) + wk * l;
        lacc[b] = acc;
        divb[b] = 0.f;                                 // ready for next RHS
        if (stage == 3) lp[b] += dt6 * acc;
    }
}

__global__ void bpd_kernel(const float* __restrict__ z, const float* __restrict__ lp,
                           float* __restrict__ out) {
    __shared__ float red[256];
    int b = blockIdx.x;
    float s = 0.f;
    for (int i = threadIdx.x; i < ND; i += 256) {
        float v = z[(size_t)b * ND + i];
        s += v * v;
    }
    red[threadIdx.x] = s;
    __syncthreads();
    for (int o = 128; o > 0; o >>= 1) {
        if (threadIdx.x < o) red[threadIdx.x] += red[threadIdx.x + o];
        __syncthreads();
    }
    if (threadIdx.x == 0) {
        float N = (float)ND;
        // bpd = (0.5*N*ln(2pi) + 0.5*sum(z^2) - delta_logp) / (ln2 * N)
        out[b] = (0.5f * N * 1.837877066409345f + 0.5f * red[0] - lp[b])
                 / (0.6931471805599453f * N);
    }
}

// ---------------- host driver ----------------

extern "C" void kernel_launch(void* const* d_in, const int* in_sizes, int n_in,
                              void* d_out, int out_size, void* d_ws, size_t ws_size,
                              hipStream_t stream) {
    (void)in_sizes; (void)n_in; (void)out_size; (void)ws_size;
    const float* patches = (const float*)d_in[0];
    const float* eps     = (const float*)d_in[1];
    const float* w1      = (const float*)d_in[2];
    const float* b1      = (const float*)d_in[3];
    const float* wt      = (const float*)d_in[4];
    const float* bt      = (const float*)d_in[5];
    const float* w2      = (const float*)d_in[6];
    const float* b2      = (const float*)d_in[7];
    const float* w3      = (const float*)d_in[8];
    const float* b3      = (const float*)d_in[9];
    float* out = (float*)d_out;

    char* ws = (char*)d_ws;
    size_t off = 0;
    auto alloc = [&](size_t bytes) -> char* {
        char* p = ws + off;
        off = (off + bytes + 255) & ~(size_t)255;
        return p;
    };
    float*    xcur = (float*)alloc((size_t)NB * ND * 4);
    float*    xt   = (float*)alloc((size_t)NB * ND * 4);
    float*    kbuf = (float*)alloc((size_t)NB * ND * 4);
    float*    kacc = (float*)alloc((size_t)NB * ND * 4);
    _Float16* h1   = (_Float16*)alloc((size_t)NB * NPIX * 32 * 2);
    _Float16* th1  = (_Float16*)alloc((size_t)NB * NPIX * 32 * 2);
    _Float16* h2   = (_Float16*)alloc((size_t)NB * NPIX * 32 * 2);
    _Float16* th2  = (_Float16*)alloc((size_t)NB * NPIX * 32 * 2);
    _Float16* u1   = (_Float16*)alloc((size_t)NPIX * 32 * 2);
    v16h*     pw1  = (v16h*)alloc(2 * 32 * sizeof(v16h));
    v16h*     pw2  = (v16h*)alloc(2 * 9 * 32 * sizeof(v16h));
    v16h*     pw3  = (v16h*)alloc(9 * 32 * sizeof(v16h));
    float*    embb = (float*)alloc(40 * 32 * 4);
    float*    divb = (float*)alloc(256);
    float*    lacc = (float*)alloc(256);
    float*    lp   = (float*)alloc(256);

    const float dt = (1.0f - T_EPS) / 10.0f;

    init_kernel<<<(NB * ND + 255) / 256, 256, 0, stream>>>(patches, xcur, lp, lacc, divb);
    emb_kernel<<<5, 256, 0, stream>>>(wt, bt, b1, embb, dt);
    pack_w1_kernel<<<1, 64, 0, stream>>>(w1, pw1);
    pack_w2_kernel<<<1, 576, 0, stream>>>(w2, pw2);
    pack_w3_kernel<<<1, 288, 0, stream>>>(w3, pw3);
    u1_kernel<<<(NPIX * 32 + 255) / 256, 256, 0, stream>>>(eps, w1, u1);

    const double ca[4]   = {0.0, 0.5, 0.5, 1.0};
    const float  wks[4]  = {1.f, 2.f, 2.f, 1.f};
    const float  cdts[4] = {0.5f * dt, 0.5f * dt, dt, 0.f};
    const float  dt6 = dt / 6.f;

    for (int s = 0; s < 10; ++s) {
        double t0 = (double)T_EPS + (double)dt * s;
        for (int a = 0; a < 4; ++a) {
            double t    = t0 + ca[a] * (double)dt;
            double beta = 0.1 + t * 19.9;
            double lmc  = -0.25 * t * t * 19.9 - 0.5 * t * 0.1;
            double stdv = 1.0 - exp(2.0 * lmc);
            double g2   = beta * (1.0 - exp(4.0 * lmc));
            float  cc   = (float)(0.5 * g2 / stdv);
            float  betaf = (float)beta;
            const float* xin = (a == 0) ? xcur : xt;
            int eidx = s * 4 + a;

            conv1_kernel<<<NTILES / 8, 256, 0, stream>>>(xin, h1, th1, u1, pw1, embb, eidx);
            conv2_kernel<<<NTILES / 8, 256, 0, stream>>>(h1, th1, h2, th2, pw2, b2);
            conv3_drift_kernel<<<NTILES / 8, 256, 0, stream>>>(h2, th2, xin, eps, pw3, b3,
                                                               kbuf, divb, betaf, cc);
            scalar_kernel<<<1, 32, 0, stream>>>(divb, lacc, lp, betaf, wks[a], a, dt6);
            float* xo = (a < 3) ? xt : xcur;
            combine_kernel<<<(NB * ND + 255) / 256, 256, 0, stream>>>(xcur, kbuf, kacc, xo,
                                                                      wks[a], cdts[a], a, dt6);
        }
    }
    bpd_kernel<<<NB, 256, 0, stream>>>(xcur, lp, out);
}
